// FeatureFusionModule_50637664420272
// MI455X (gfx1250) — matrix operand order repaired and checked by reading, not compiled, not run on hardware
//
#include <hip/hip_runtime.h>
#include <math.h>

// ---------------------------------------------------------------------------
// CDNA5 (gfx1250) implementation. wave32. GEMM stages use
// v_wmma_f32_16x16x32_bf16 (fp32 accumulate). hi_res is repacked into a
// zero-padded NHWC bf16 buffer so every WMMA B-fragment is a single
// contiguous 32B load (no bounds checks, no divergence). All hot buffers
// are L2-resident (192MB L2).
// ---------------------------------------------------------------------------

typedef __bf16 bf16_t;
typedef __attribute__((ext_vector_type(16))) __bf16 v16bf;
typedef __attribute__((ext_vector_type(8)))  float  v8f;

#define BATCH 2
#define CHI   1024
#define CLO   2048
#define HY    64
#define WX    64
#define HW    (HY*WX)
#define HP    66          // padded spatial (1-px halo for 3x3 conv)
#define HLO   32
#define WLO   32
#define HWLO  (HLO*WLO)
#define NP    9
#define GDCN  32
#define GPOST 256
#define COUT  2048

#define KC_OFF 288   // offset conv K-chunks: 9 taps * (1024/32) channel-chunks
#define KC_DCN 9     // dcn K-chunks: 9 taps, 32 group-channels each

// ---- workspace layout (bytes) ----
#define OFF_BUF_OFF   0
#define OFF_BUF_SZ    (BATCH*27*HW*4)                 //    884,736
#define HIAL_OFF      (OFF_BUF_OFF + OFF_BUF_SZ)
#define HIAL_SZ       (BATCH*CHI*HW*4)                // 33,554,432
#define HIBP_OFF      (HIAL_OFF + HIAL_SZ)
#define HIBP_SZ       (BATCH*HP*HP*CHI*2)             // 17,842,176 (NHWC, padded)
#define AOFF_OFF      (HIBP_OFF + HIBP_SZ)            // 32B aligned
#define AOFF_SZ       (2*KC_OFF*32*16*2)              //    589,824
#define ADCN_OFF      (AOFF_OFF + AOFF_SZ)
#define ADCN_SZ       (GDCN*2*KC_DCN*32*16*2)         //    589,824
// total ~53.5 MB

// Per-lane K index inside a 32-wide chunk for the 16x32 bf16 A fragment
// (contiguous, matching B): lane<16 holds K {0..15}; lane>=16 holds K {16..31}.
__device__ __forceinline__ int a_klocal(int lane, int j) {
  return ((lane < 16) ? 0 : 16) + j;
}

// ---------------------------------------------------------------------------
// K0a: hi_res (B,C,H,W) fp32 -> zero-padded NHWC bf16 (B, 66, 66, 1024).
// Consecutive threads walk channels -> fully coalesced 64B stores; strided
// reads are absorbed by L2 (hi_res read exactly once).
__global__ void k_pack_hi_nhwc(const float* __restrict__ hi,
                               bf16_t* __restrict__ hib) {
  int idx = blockIdx.x * 256 + threadIdx.x;      // B*66*66*1024 total
  int c  = idx & 1023;
  int t  = idx >> 10;
  int x  = t % HP;
  int yz = t / HP;
  int y  = yz % HP;
  int b  = yz / HP;
  bf16_t v = (bf16_t)0.0f;
  if (y >= 1 && y <= HY && x >= 1 && x <= WX)
    v = (bf16_t)hi[((size_t)(b * CHI + c)) * HW + (y - 1) * WX + (x - 1)];
  hib[idx] = v;
}

// ---------------------------------------------------------------------------
// K0b: swizzle w_off (27,1024,3,3) -> padded A fragments [mt][kc][lane][16]
__global__ void k_swz_woff(const float* __restrict__ w, bf16_t* __restrict__ a) {
  int idx = blockIdx.x * blockDim.x + threadIdx.x;
  if (idx >= 2 * KC_OFF * 512) return;
  int j    = idx & 15;
  int lane = (idx >> 4) & 31;
  int kc   = (idx >> 9) % KC_OFF;
  int mt   = idx / (KC_OFF * 512);
  int kl   = a_klocal(lane, j);
  int m    = mt * 16 + (lane & 15);
  int tap  = kc >> 5;                 // 0..8
  int c    = (kc & 31) * 32 + kl;     // input channel
  a[idx] = (m < 27) ? (bf16_t)w[(m * CHI + c) * NP + tap] : (bf16_t)0.0f;
}

// ---------------------------------------------------------------------------
// K0c: swizzle w_dcn (1024,32,3,3) -> A fragments [g][mt][kc=p][lane][16]
// DCN K order: K = p*32 + i  (tap-major, 32 group-channels per chunk)
__global__ void k_swz_wdcn(const float* __restrict__ w, bf16_t* __restrict__ a) {
  int idx = blockIdx.x * blockDim.x + threadIdx.x;
  if (idx >= GDCN * 2 * KC_DCN * 512) return;
  int j    = idx & 15;
  int lane = (idx >> 4) & 31;
  int kc   = (idx >> 9) % KC_DCN;           // tap p
  int mt   = (idx / (KC_DCN * 512)) & 1;
  int g    = idx / (2 * KC_DCN * 512);
  int i    = a_klocal(lane, j);             // group-channel 0..31
  int o    = mt * 16 + (lane & 15);
  a[idx] = (bf16_t)w[(((size_t)(g * 32 + o)) * 32 + i) * NP + kc];
}

// ---------------------------------------------------------------------------
// K1: offset conv, implicit-im2col WMMA GEMM. M=27(pad32), K=9216, N=8192.
// 1 wave per 16-pixel row segment; 9 taps x 32 channel-chunks, 2 M-tiles
// -> 576 wmma/wave. B-fragment = one contiguous 32B load from padded NHWC.
__global__ __launch_bounds__(256) void k_off_conv(
    const bf16_t* __restrict__ hib, const bf16_t* __restrict__ awz,
    const float* __restrict__ boff, float* __restrict__ off_buf) {
  int lane = threadIdx.x & 31;
  int tile = blockIdx.x * 8 + (threadIdx.x >> 5);   // 0..511
  int b    = tile >> 8;
  int rem  = tile & 255;
  int y    = rem >> 2;
  int xb   = (rem & 3) << 4;
  int x    = xb + (lane & 15);
  int half = (lane < 16) ? 0 : 16;

  v8f acc0 = {0.f, 0.f, 0.f, 0.f, 0.f, 0.f, 0.f, 0.f};
  v8f acc1 = acc0;
  const bf16_t* abase0 = awz + (size_t)lane * 16;
  const bf16_t* abase1 = awz + ((size_t)KC_OFF * 512) + (size_t)lane * 16;

#pragma unroll
  for (int tap = 0; tap < NP; ++tap) {
    const int ky = tap / 3 - 1, kx = tap % 3 - 1;
    // padded coords: +1 halo offset
    const bf16_t* prow = hib +
        (((size_t)b * HP + (y + ky + 1)) * HP + (x + kx + 1)) * CHI + half;
    const bf16_t* pa0 = abase0 + (size_t)tap * 32 * 512;
    const bf16_t* pa1 = abase1 + (size_t)tap * 32 * 512;
#pragma unroll 4
    for (int cc = 0; cc < 32; ++cc) {
      v16bf bfr = *(const v16bf*)(prow + cc * 32);
      v16bf a0  = *(const v16bf*)(pa0 + (size_t)cc * 512);
      v16bf a1  = *(const v16bf*)(pa1 + (size_t)cc * 512);
      acc0 = __builtin_amdgcn_wmma_f32_16x16x32_bf16(false, a0, false, bfr,
                                                     (short)0, acc0, false, false);
      acc1 = __builtin_amdgcn_wmma_f32_16x16x32_bf16(false, a1, false, bfr,
                                                     (short)0, acc1, false, false);
    }
  }
  // D layout: lane<16 -> N=lane, rows v; lane>=16 -> N=lane-16, rows v+8
#pragma unroll
  for (int v = 0; v < 8; ++v) {
    int mrow = v + ((lane < 16) ? 0 : 8);
    int px = y * WX + xb + (lane & 15);
    int m0 = mrow;
    int m1 = 16 + mrow;
    if (m0 < 27) off_buf[((size_t)(b * 27 + m0)) * HW + px] = acc0[v] + boff[m0];
    if (m1 < 27) off_buf[((size_t)(b * 27 + m1)) * HW + px] = acc1[v] + boff[m1];
  }
}

// ---------------------------------------------------------------------------
// K2: fused deformable sampling + grouped WMMA GEMM.
// One block = (batch b, group g, image row y): phase 1 precomputes clamped
// corner indices + ok-masked modulated bilinear weights; phase 2 builds
// V (288 x 64) bf16 in LDS with 4 unconditional loads + 4 FMA per element;
// phase 3: 4 waves x (2 Mtiles x 9 Kchunks) wmma.
#define PIXB 64
#define VSTR 304   // padded row stride (elems): 608B rows -> 32B-aligned frags
__global__ __launch_bounds__(128) void k_dcn(
    const float* __restrict__ hi, const float* __restrict__ off_buf,
    const bf16_t* __restrict__ adcn, const float* __restrict__ bdcn,
    float* __restrict__ hial) {
  __shared__ int   s_i00[NP * PIXB];
  __shared__ int   s_i01[NP * PIXB];
  __shared__ int   s_i10[NP * PIXB];
  __shared__ int   s_i11[NP * PIXB];
  __shared__ float s_w00[NP * PIXB];
  __shared__ float s_w01[NP * PIXB];
  __shared__ float s_w10[NP * PIXB];
  __shared__ float s_w11[NP * PIXB];
  __shared__ __align__(32) bf16_t Vlds[PIXB * VSTR];

  int bid = blockIdx.x;
  int y = bid & 63;
  int g = (bid >> 6) & 31;
  int b = bid >> 11;
  int tid = threadIdx.x;

  // Phase 1: sampling geometry (reference semantics: clip index + ok mask)
  for (int it = tid; it < NP * PIXB; it += 128) {
    int xc = it & 63;
    int p  = it >> 6;
    int pix = y * WX + xc;
    float dy = off_buf[((size_t)(b * 27 + p)) * HW + pix];
    float dx = off_buf[((size_t)(b * 27 + 9 + p)) * HW + pix];
    float mr = off_buf[((size_t)(b * 27 + 18 + p)) * HW + pix];
    float mk = 1.0f / (1.0f + expf(-mr));
    float sy = (float)(y + p / 3 - 1) + dy;
    float sx = (float)(xc + p % 3 - 1) + dx;
    float y0f = floorf(sy), x0f = floorf(sx);
    float wy = sy - y0f,  wx = sx - x0f;
    int y0 = (int)y0f, x0 = (int)x0f;
    int y1 = y0 + 1,   x1 = x0 + 1;
    float oky0 = (y0 >= 0 && y0 < HY) ? 1.f : 0.f;
    float oky1 = (y1 >= 0 && y1 < HY) ? 1.f : 0.f;
    float okx0 = (x0 >= 0 && x0 < WX) ? 1.f : 0.f;
    float okx1 = (x1 >= 0 && x1 < WX) ? 1.f : 0.f;
    int yc0 = min(max(y0, 0), HY - 1), yc1 = min(max(y1, 0), HY - 1);
    int xc0 = min(max(x0, 0), WX - 1), xc1 = min(max(x1, 0), WX - 1);
    s_i00[it] = yc0 * WX + xc0;
    s_i01[it] = yc0 * WX + xc1;
    s_i10[it] = yc1 * WX + xc0;
    s_i11[it] = yc1 * WX + xc1;
    s_w00[it] = mk * (1.f - wy) * (1.f - wx) * oky0 * okx0;
    s_w01[it] = mk * (1.f - wy) * wx         * oky0 * okx1;
    s_w10[it] = mk * wy * (1.f - wx)         * oky1 * okx0;
    s_w11[it] = mk * wy * wx                 * oky1 * okx1;
  }
  __syncthreads();

  // Phase 2: build modulated V[k = p*32+i][n] in LDS (bf16)
  for (int it = tid; it < 288 * PIXB; it += 128) {
    int n = it & 63;
    int k = it >> 6;
    int p = k >> 5;
    int i = k & 31;
    int ti = p * 64 + n;
    const float* pl = hi + ((size_t)(b * CHI + g * 32 + i)) * HW;
    float v = s_w00[ti] * pl[s_i00[ti]] + s_w01[ti] * pl[s_i01[ti]] +
              s_w10[ti] * pl[s_i10[ti]] + s_w11[ti] * pl[s_i11[ti]];
    Vlds[n * VSTR + k] = (bf16_t)v;
  }
  __syncthreads();

  // Phase 3: WMMA. Wave w handles N-tile of 16 pixels; M=32 (2 tiles), K=288.
  int lane = tid & 31;
  int wv   = tid >> 5;
  int half = (lane < 16) ? 0 : 16;
  v8f acc0 = {0.f, 0.f, 0.f, 0.f, 0.f, 0.f, 0.f, 0.f};
  v8f acc1 = acc0;
#pragma unroll
  for (int kc = 0; kc < KC_DCN; ++kc) {
    v16bf a0 = *(const v16bf*)(adcn + ((size_t)((g * 2 + 0) * KC_DCN + kc) * 32 + lane) * 16);
    v16bf a1 = *(const v16bf*)(adcn + ((size_t)((g * 2 + 1) * KC_DCN + kc) * 32 + lane) * 16);
    v16bf bfr = *(const v16bf*)(&Vlds[(wv * 16 + (lane & 15)) * VSTR + kc * 32 + half]);
    acc0 = __builtin_amdgcn_wmma_f32_16x16x32_bf16(false, a0, false, bfr,
                                                   (short)0, acc0, false, false);
    acc1 = __builtin_amdgcn_wmma_f32_16x16x32_bf16(false, a1, false, bfr,
                                                   (short)0, acc1, false, false);
  }
#pragma unroll
  for (int v = 0; v < 8; ++v) {
    int orow = v + ((lane < 16) ? 0 : 8);
    int xc = wv * 16 + (lane & 15);
    int c0 = g * 32 + orow;
    int c1 = g * 32 + 16 + orow;
    size_t pix = (size_t)y * WX + xc;
    hial[((size_t)(b * CHI + c0)) * HW + pix] = acc0[v] + bdcn[c0];
    hial[((size_t)(b * CHI + c1)) * HW + pix] = acc1[v] + bdcn[c1];
  }
}

// ---------------------------------------------------------------------------
// K3: grouped 1x1 post conv (8x12 per group, too small for WMMA) with the
// bilinear lo-upsample computed inline (saves 134MB of lo_up traffic).
__global__ __launch_bounds__(256) void k_post(
    const float* __restrict__ lo, const float* __restrict__ hial,
    const float* __restrict__ wp, const float* __restrict__ bpo,
    float* __restrict__ out) {
  int t = blockIdx.x * 256 + threadIdx.x;   // (b, g, pix)
  int pix = t & 4095;
  int g = (t >> 12) & 255;
  int b = t >> 20;
  int y = pix >> 6, x = pix & 63;
  // half-pixel-center bilinear, scale 2 (jax.image.resize 'bilinear')
  float sy = y * 0.5f - 0.25f, sx = x * 0.5f - 0.25f;
  float y0f = floorf(sy), x0f = floorf(sx);
  float fy = sy - y0f, fx = sx - x0f;
  int iy0 = min(max((int)y0f, 0), HLO - 1);
  int iy1 = min(max((int)y0f + 1, 0), HLO - 1);
  int ix0 = min(max((int)x0f, 0), WLO - 1);
  int ix1 = min(max((int)x0f + 1, 0), WLO - 1);
  float w00 = (1.f - fy) * (1.f - fx), w01 = (1.f - fy) * fx;
  float w10 = fy * (1.f - fx),         w11 = fy * fx;

  float xin[12];
#pragma unroll
  for (int i = 0; i < 12; ++i) {
    int c = g * 12 + i;
    float v;
    if (c < CLO) {
      const float* pl = lo + ((size_t)(b * CLO + c)) * HWLO;
      v = w00 * pl[iy0 * WLO + ix0] + w01 * pl[iy0 * WLO + ix1] +
          w10 * pl[iy1 * WLO + ix0] + w11 * pl[iy1 * WLO + ix1];
    } else {
      v = hial[((size_t)(b * CHI + (c - CLO))) * HW + pix];
    }
    xin[i] = v;
  }
#pragma unroll
  for (int o = 0; o < 8; ++o) {
    int co = g * 8 + o;
    float acc = bpo[co];
#pragma unroll
    for (int i = 0; i < 12; ++i) acc += wp[co * 12 + i] * xin[i];
    out[((size_t)(b * COUT + co)) * HW + pix] = acc;
  }
}

// ---------------------------------------------------------------------------
extern "C" void kernel_launch(void* const* d_in, const int* in_sizes, int n_in,
                              void* d_out, int out_size, void* d_ws, size_t ws_size,
                              hipStream_t stream) {
  (void)in_sizes; (void)n_in; (void)out_size; (void)ws_size;
  const float* lo    = (const float*)d_in[0];
  const float* hi    = (const float*)d_in[1];
  const float* woff  = (const float*)d_in[2];
  const float* boff  = (const float*)d_in[3];
  const float* wdcn  = (const float*)d_in[4];
  const float* bdcn  = (const float*)d_in[5];
  const float* wpost = (const float*)d_in[6];
  const float* bpost = (const float*)d_in[7];
  float* out = (float*)d_out;

  char* ws = (char*)d_ws;
  float*  off_buf = (float*)(ws + OFF_BUF_OFF);
  float*  hial    = (float*)(ws + HIAL_OFF);
  bf16_t* hib     = (bf16_t*)(ws + HIBP_OFF);
  bf16_t* aoff    = (bf16_t*)(ws + AOFF_OFF);
  bf16_t* adcn    = (bf16_t*)(ws + ADCN_OFF);

  k_pack_hi_nhwc<<<(BATCH * HP * HP * CHI) / 256, 256, 0, stream>>>(hi, hib);
  k_swz_woff<<<(2 * KC_OFF * 512 + 255) / 256, 256, 0, stream>>>(woff, aoff);
  k_swz_wdcn<<<(GDCN * 2 * KC_DCN * 512 + 255) / 256, 256, 0, stream>>>(wdcn, adcn);
  k_off_conv<<<64, 256, 0, stream>>>(hib, aoff, boff, off_buf);
  k_dcn<<<BATCH * GDCN * HY, 128, 0, stream>>>(hi, off_buf, adcn, bdcn, hial);
  k_post<<<(BATCH * GPOST * HW) / 256, 256, 0, stream>>>(lo, hial, wpost, bpost, out);
}
